// InvariantPointAttention_55284819034663
// MI455X (gfx1250) — matrix operand order repaired and checked by reading, not compiled, not run on hardware
//
#include <hip/hip_runtime.h>
#include <hip/hip_bf16.h>
#include <math.h>

// ---- problem constants (from reference) ----
#define LQ    768
#define CS    384
#define CZ    128
#define CH    16
#define NH    12
#define PQ    4
#define PV    8
#define NBR   30
#define NPROJ 1152            // 192 (q) + 384 (kv) + 144 (qp) + 432 (kvp)
#define NCAT  2112            // 192 + 96*3 + 96 + 1536
#define NV    64              // padded [V | V_pts] columns (40 used)
#define INF_  100000.0f
#define EPS_  1e-8f

#define S1_QK  0.14433756729740643f   // sqrt(1/48)
#define S1_HW  0.13608276348795434f   // sqrt(1/54)
#define S1_B   0.57735026918962576f   // sqrt(1/3)

typedef __attribute__((ext_vector_type(16))) __bf16 v16bf;
typedef __attribute__((ext_vector_type(8)))  __bf16 v8bf;
typedef __attribute__((ext_vector_type(8)))  float  v8f;

__device__ __forceinline__ float softplusf(float x) {
    return (x > 20.f) ? x : log1pf(expf(x));
}

// ============================================================================
// Batched bf16 GEMM with 2x2 register tiling:
//   C[b] = A[b] (MxK) * B[b] (NxK)^T  (+ bias[n])
// A, B row-major bf16 (B pre-transposed so K is contiguous for both).
// One wave per 32x32 C tile: 2 A-frags + 2 B-frags -> 4 WMMAs per K-step
// (32 B/lane/WMMA instead of 64, and 4 independent accumulation chains).
// M, N multiples of 32; K multiple of 32.
// ============================================================================
__global__ void k_gemm_bf16(const __bf16* __restrict__ A,
                            const __bf16* __restrict__ B,
                            float* __restrict__ C,
                            const float* __restrict__ bias,
                            int M, int N, int K,
                            long long sA, long long sB, long long sC)
{
    const int lane   = threadIdx.x & 31;
    const int wave   = threadIdx.x >> 5;
    const int tilesN = N >> 5;               // 32-wide super-tiles
    const int tilesM = M >> 5;
    const int tile   = blockIdx.x * (blockDim.x >> 5) + wave;
    if (tile >= tilesM * tilesN) return;
    const int tm   = (tile / tilesN) << 5;
    const int tn   = (tile % tilesN) << 5;
    const int row  = lane & 15;   // A-row / B-col / C-col within 16x16 sub-tile
    const int half = lane >> 4;   // K-half selector

    const __bf16* Ab = A + (long long)blockIdx.y * sA;
    const __bf16* Bb = B + (long long)blockIdx.y * sB;
    float*        Cb = C + (long long)blockIdx.y * sC;

    const __bf16* ap0 = Ab + (long long)(tm + row) * K;
    const __bf16* ap1 = ap0 + (long long)16 * K;
    const __bf16* bp0 = Bb + (long long)(tn + row) * K;
    const __bf16* bp1 = bp0 + (long long)16 * K;

    v8f acc00 = {}, acc01 = {}, acc10 = {}, acc11 = {};

    for (int k0 = 0; k0 < K; k0 += 32) {
        // A 16x32 fragment: e<8 -> K = k0+8*half+e ; e>=8 -> K = k0+16+8*half+(e-8)
        v8bf a0lo = *(const v8bf*)(ap0 + k0 + 8 * half);
        v8bf a0hi = *(const v8bf*)(ap0 + k0 + 16 + 8 * half);
        v8bf a1lo = *(const v8bf*)(ap1 + k0 + 8 * half);
        v8bf a1hi = *(const v8bf*)(ap1 + k0 + 16 + 8 * half);
        // B 32x16 fragment (stored transposed): e<8 -> K = k0+8*half+e ; e>=8 -> K = k0+16+8*half+(e-8)
        v8bf b0lo = *(const v8bf*)(bp0 + k0 + 8 * half);
        v8bf b0hi = *(const v8bf*)(bp0 + k0 + 16 + 8 * half);
        v8bf b1lo = *(const v8bf*)(bp1 + k0 + 8 * half);
        v8bf b1hi = *(const v8bf*)(bp1 + k0 + 16 + 8 * half);
        v16bf a0, a1, b0, b1;
#pragma unroll
        for (int e = 0; e < 8; ++e) {
            a0[e] = a0lo[e]; a0[e + 8] = a0hi[e];
            a1[e] = a1lo[e]; a1[e + 8] = a1hi[e];
            b0[e] = b0lo[e]; b0[e + 8] = b0hi[e];
            b1[e] = b1lo[e]; b1[e + 8] = b1hi[e];
        }
        acc00 = __builtin_amdgcn_wmma_f32_16x16x32_bf16(false, a0, false, b0,
                                                        (short)0, acc00, false, false);
        acc01 = __builtin_amdgcn_wmma_f32_16x16x32_bf16(false, a0, false, b1,
                                                        (short)0, acc01, false, false);
        acc10 = __builtin_amdgcn_wmma_f32_16x16x32_bf16(false, a1, false, b0,
                                                        (short)0, acc10, false, false);
        acc11 = __builtin_amdgcn_wmma_f32_16x16x32_bf16(false, a1, false, b1,
                                                        (short)0, acc11, false, false);
    }

    // C: col = tn + ni*16 + row, row r of acc -> M = tm + mi*16 + r + 8*half
    const int n0 = tn + row, n1 = tn + 16 + row;
    const float bad0 = bias ? bias[n0] : 0.f;
    const float bad1 = bias ? bias[n1] : 0.f;
#pragma unroll
    for (int r = 0; r < 8; ++r) {
        const int m0 = tm + r + 8 * half;
        const int m1 = m0 + 16;
        Cb[(long long)m0 * N + n0] = acc00[r] + bad0;
        Cb[(long long)m0 * N + n1] = acc01[r] + bad1;
        Cb[(long long)m1 * N + n0] = acc10[r] + bad0;
        Cb[(long long)m1 * N + n1] = acc11[r] + bad1;
    }
}

// ---------------------------------------------------------------------------
__global__ void k_f2bf(const float* __restrict__ x, __bf16* __restrict__ y, int n)
{
    int i = blockIdx.x * blockDim.x + threadIdx.x;
    if (i < n) y[i] = (__bf16)x[i];
}

// Pack the four projection weight matrices into one transposed bf16 (NPROJ x CS)
// matrix, and concatenate biases.
__global__ void k_pack_w(const float* __restrict__ Wq,  const float* __restrict__ Wkv,
                         const float* __restrict__ Wqp, const float* __restrict__ Wkvp,
                         const float* __restrict__ bq,  const float* __restrict__ bkv,
                         const float* __restrict__ bqp, const float* __restrict__ bkvp,
                         __bf16* __restrict__ Wt, float* __restrict__ ball)
{
    int idx = blockIdx.x * blockDim.x + threadIdx.x;
    if (idx < NPROJ) {
        int c = idx; float bv;
        if      (c < 192) bv = bq[c];
        else if (c < 576) bv = bkv[c - 192];
        else if (c < 720) bv = bqp[c - 576];
        else              bv = bkvp[c - 720];
        ball[c] = bv;
    }
    if (idx >= NPROJ * CS) return;
    int col = idx / CS, k = idx % CS;
    float w;
    if      (col < 192) w = Wq [k * 192 + col];
    else if (col < 576) w = Wkv[k * 384 + (col - 192)];
    else if (col < 720) w = Wqp[k * 144 + (col - 576)];
    else                w = Wkvp[k * 432 + (col - 720)];
    Wt[(long long)col * CS + k] = (__bf16)w;
}

// Transpose W_out (NCAT x CS) -> bf16 (CS x NCAT)
__global__ void k_pack_wout(const float* __restrict__ W, __bf16* __restrict__ Wt)
{
    int idx = blockIdx.x * blockDim.x + threadIdx.x;
    if (idx >= CS * NCAT) return;
    int n = idx / NCAT, k = idx % NCAT;
    Wt[idx] = (__bf16)W[(long long)k * CS + n];
}

// Apply rigid transforms to point projections: q_pts, k_pts, v_pts (f32).
__global__ void k_points(const float* __restrict__ proj, const float* __restrict__ rot,
                         const float* __restrict__ trans,
                         float* __restrict__ qpts, float* __restrict__ kpts,
                         float* __restrict__ vpts)
{
    const int PT = 2 * PQ + PV; // 16 point slots per (i,h)
    int idx = blockIdx.x * blockDim.x + threadIdx.x;
    if (idx >= LQ * NH * PT) return;
    int p = idx % PT; int t = idx / PT;
    int h = t % NH;   int i = t / NH;
    const float* pr = proj + (long long)i * NPROJ;
    float f[3];
    if (p < PQ) {
#pragma unroll
        for (int j = 0; j < 3; ++j) f[j] = pr[576 + j * 48 + h * PQ + p];
    } else {
        int pp = p - PQ; // 0..11 within kv points
#pragma unroll
        for (int j = 0; j < 3; ++j) f[j] = pr[720 + j * 144 + h * 12 + pp];
    }
    const float* R = rot + (long long)i * 9;
    const float* T = trans + (long long)i * 3;
    float o0 = R[0]*f[0] + R[1]*f[1] + R[2]*f[2] + T[0];
    float o1 = R[3]*f[0] + R[4]*f[1] + R[5]*f[2] + T[1];
    float o2 = R[6]*f[0] + R[7]*f[1] + R[8]*f[2] + T[2];
    if (p < PQ) {
        float* o = qpts + (((long long)i * NH + h) * PQ + p) * 3;
        o[0] = o0; o[1] = o1; o[2] = o2;
    } else if (p < 2 * PQ) {
        float* o = kpts + (((long long)i * NH + h) * PQ + (p - PQ)) * 3;
        o[0] = o0; o[1] = o1; o[2] = o2;
    } else {
        float* o = vpts + (((long long)i * NH + h) * PV + (p - 2 * PQ)) * 3;
        o[0] = o0; o[1] = o1; o[2] = o2;
    }
}

// Per-(i,h) squared norms of q_pts / k_pts (for |q-k|^2 expansion).
__global__ void k_ptsq(const float* __restrict__ qpts, const float* __restrict__ kpts,
                       float* __restrict__ qq, float* __restrict__ kk)
{
    int idx = blockIdx.x * blockDim.x + threadIdx.x;
    if (idx >= LQ * NH) return;
    const float* qp = qpts + (long long)idx * PQ * 3;
    const float* kp = kpts + (long long)idx * PQ * 3;
    float sq = 0.f, sk = 0.f;
#pragma unroll
    for (int r = 0; r < PQ * 3; ++r) { sq += qp[r]*qp[r]; sk += kp[r]*kp[r]; }
    qq[idx] = sq; kk[idx] = sk;
}

// Augmented bf16 logit operands: 32 cols = [16 q*sqrt(1/48) | 12 q_pts*hw_h | 0 0 0 0]
__global__ void k_build_aug(const float* __restrict__ proj,
                            const float* __restrict__ qpts, const float* __restrict__ kpts,
                            const float* __restrict__ hwraw,
                            __bf16* __restrict__ qa, __bf16* __restrict__ ka)
{
    int idx = blockIdx.x * blockDim.x + threadIdx.x;
    if (idx >= NH * LQ * 32) return;
    int e = idx & 31; int t = idx >> 5;
    int i = t % LQ;   int h = t / LQ;
    float hw = softplusf(hwraw[h]) * S1_HW;
    float qv = 0.f, kv = 0.f;
    if (e < 16) {
        qv = proj[(long long)i * NPROJ + h * CH + e] * S1_QK;
        kv = proj[(long long)i * NPROJ + 192 + h * 32 + e];
    } else if (e < 28) {
        int r = e - 16;
        qv = qpts[((long long)i * NH + h) * PQ * 3 + r] * hw;
        kv = kpts[((long long)i * NH + h) * PQ * 3 + r];
    }
    long long o = (long long)h * LQ * 32 + (long long)i * 32 + e;
    qa[o] = (__bf16)qv;
    ka[o] = (__bf16)kv;
}

// Sparse pair bias: bias_sp(i,k,h) = sum_c z(i,k,c) * W_b(c,h)
__global__ void k_bias_sp(const float* __restrict__ z, const float* __restrict__ Wb,
                          float* __restrict__ bsp)
{
    int idx = blockIdx.x * blockDim.x + threadIdx.x;
    if (idx >= LQ * NBR * NH) return;
    int h = idx % NH; int t = idx / NH;
    const float* zr = z + (long long)t * CZ;
    float acc = 0.f;
    for (int c = 0; c < CZ; ++c) acc += zr[c] * Wb[c * NH + h];
    bsp[idx] = acc;
}

// Elementwise logit completion: rank-1 point terms, b_b, mask.
__global__ void k_finalize(float* __restrict__ logits,
                           const float* __restrict__ qq, const float* __restrict__ kk,
                           const float* __restrict__ bb, const float* __restrict__ hwraw,
                           const float* __restrict__ mask)
{
    long long idx = (long long)blockIdx.x * blockDim.x + threadIdx.x;
    if (idx >= (long long)NH * LQ * LQ) return;
    int j = (int)(idx % LQ); long long t = idx / LQ;
    int i = (int)(t % LQ);   int h = (int)(t / LQ);
    float hw = softplusf(hwraw[h]) * S1_HW;
    float add = -0.5f * hw * (qq[i * NH + h] + kk[j * NH + h])
              + S1_B * bb[h]
              + INF_ * (mask[i] * mask[j] - 1.f);
    logits[idx] += add;
}

// Scatter sparse bias into dense logits (E_idx entries are distinct per row).
__global__ void k_scatter(float* __restrict__ logits, const float* __restrict__ bsp,
                          const int* __restrict__ E)
{
    int idx = blockIdx.x * blockDim.x + threadIdx.x;
    if (idx >= LQ * NBR * NH) return;
    int h = idx % NH; int t = idx / NH;
    int k = t % NBR;  int i = t / NBR;
    int j = E[i * NBR + k];
    logits[(long long)h * LQ * LQ + (long long)i * LQ + j] += S1_B * bsp[idx];
}

// Row softmax (f32 logits -> bf16 probabilities). One block per (h,i) row.
__global__ void k_softmax(const float* __restrict__ logits, __bf16* __restrict__ probs)
{
    __shared__ float red[256];
    const int t = threadIdx.x;
    const long long rowoff = (long long)blockIdx.x * LQ;
    const float* lr = logits + rowoff;
    float m = -3.4e38f;
    for (int j = t; j < LQ; j += 256) m = fmaxf(m, lr[j]);
    red[t] = m; __syncthreads();
    for (int s = 128; s > 0; s >>= 1) { if (t < s) red[t] = fmaxf(red[t], red[t + s]); __syncthreads(); }
    m = red[0]; __syncthreads();
    float sum = 0.f;
    for (int j = t; j < LQ; j += 256) sum += __expf(lr[j] - m);
    red[t] = sum; __syncthreads();
    for (int s = 128; s > 0; s >>= 1) { if (t < s) red[t] += red[t + s]; __syncthreads(); }
    const float inv = 1.f / red[0];
    for (int j = t; j < LQ; j += 256)
        probs[rowoff + j] = (__bf16)(__expf(lr[j] - m) * inv);
}

// Pack per-head [V | V_pts | zero-pad] into bf16 (NH x NV x LQ), transposed for GEMM B.
__global__ void k_build_vcat(const float* __restrict__ proj, const float* __restrict__ vpts,
                             __bf16* __restrict__ vcat)
{
    int idx = blockIdx.x * blockDim.x + threadIdx.x;
    if (idx >= NH * NV * LQ) return;
    int j = idx % LQ; int t = idx / LQ;
    int c = t % NV;   int h = t / NV;
    float v = 0.f;
    if (c < 16)       v = proj[(long long)j * NPROJ + 192 + h * 32 + 16 + c];
    else if (c < 40)  v = vpts[((long long)j * NH + h) * PV * 3 + (c - 16)];
    vcat[idx] = (__bf16)v;
}

// Sparse o_pair: cat[i, 576 + h*128 + c] = sum_k probs(h,i,E(i,k)) * z(i,k,c)
__global__ void k_opair(const __bf16* __restrict__ probs, const float* __restrict__ z,
                        const int* __restrict__ E, __bf16* __restrict__ cat)
{
    __shared__ float as[NH * NBR];
    const int i = blockIdx.x;
    const int t = threadIdx.x; // 128 threads
    for (int u = t; u < NH * NBR; u += blockDim.x) {
        int h = u / NBR, k = u % NBR;
        int j = E[i * NBR + k];
        as[u] = (float)probs[(long long)h * LQ * LQ + (long long)i * LQ + j];
    }
    __syncthreads();
    const float* zr = z + (long long)i * NBR * CZ;
    for (int h = 0; h < NH; ++h) {
        float acc = 0.f;
#pragma unroll
        for (int k = 0; k < NBR; ++k) acc += as[h * NBR + k] * zr[k * CZ + t];
        cat[(long long)i * NCAT + 576 + h * CZ + t] = (__bf16)acc;
    }
}

// Assemble cat row: o copy, inverse-transformed o_pt (x,y,z blocks), norms.
__global__ void k_fincat(const float* __restrict__ ov, const float* __restrict__ rot,
                         const float* __restrict__ trans, __bf16* __restrict__ cat)
{
    const int i = blockIdx.x;
    const float* R = rot + (long long)i * 9;
    const float* T = trans + (long long)i * 3;
    for (int task = threadIdx.x; task < 288; task += blockDim.x) {
        if (task < 192) {
            int h = task / 16, c = task % 16;
            cat[(long long)i * NCAT + task] =
                (__bf16)ov[(long long)h * LQ * NV + (long long)i * NV + c];
        } else {
            int u = task - 192;             // 0..95 = h*PV + p
            int h = u / PV, p = u % PV;
            const float* o = ov + (long long)h * LQ * NV + (long long)i * NV + 16 + p * 3;
            float g0 = o[0] - T[0], g1 = o[1] - T[1], g2 = o[2] - T[2];
            float l0 = R[0]*g0 + R[3]*g1 + R[6]*g2;   // rot^T * g
            float l1 = R[1]*g0 + R[4]*g1 + R[7]*g2;
            float l2 = R[2]*g0 + R[5]*g1 + R[8]*g2;
            __bf16* cr = cat + (long long)i * NCAT;
            cr[192 +   0 + u] = (__bf16)l0;
            cr[192 +  96 + u] = (__bf16)l1;
            cr[192 + 192 + u] = (__bf16)l2;
            cr[480 + u] = (__bf16)sqrtf(l0*l0 + l1*l1 + l2*l2 + EPS_);
        }
    }
}

// ============================================================================
extern "C" void kernel_launch(void* const* d_in, const int* in_sizes, int n_in,
                              void* d_out, int out_size, void* d_ws, size_t ws_size,
                              hipStream_t stream)
{
    (void)in_sizes; (void)n_in; (void)out_size; (void)ws_size;

    const float* s     = (const float*)d_in[0];
    const float* z     = (const float*)d_in[1];
    const float* rot   = (const float*)d_in[2];
    const float* trans = (const float*)d_in[3];
    const float* mask  = (const float*)d_in[4];
    const int*   Eidx  = (const int*)  d_in[5];
    const float* W_q   = (const float*)d_in[6];
    const float* b_q   = (const float*)d_in[7];
    const float* W_kv  = (const float*)d_in[8];
    const float* b_kv  = (const float*)d_in[9];
    const float* W_qp  = (const float*)d_in[10];
    const float* b_qp  = (const float*)d_in[11];
    const float* W_kvp = (const float*)d_in[12];
    const float* b_kvp = (const float*)d_in[13];
    const float* W_b   = (const float*)d_in[14];
    const float* b_b   = (const float*)d_in[15];
    const float* hwts  = (const float*)d_in[16];
    const float* W_out = (const float*)d_in[17];
    const float* b_out = (const float*)d_in[18];
    float* out = (float*)d_out;

    // workspace carve-out
    char* base = (char*)d_ws;
    size_t off = 0;
    auto alloc = [&](size_t bytes) -> char* {
        char* p = base + off;
        off = (off + bytes + 255) & ~(size_t)255;
        return p;
    };
    __bf16* s_bf   = (__bf16*)alloc((size_t)LQ * CS * 2);
    __bf16* Wall_t = (__bf16*)alloc((size_t)NPROJ * CS * 2);
    float*  b_all  = (float*) alloc((size_t)NPROJ * 4);
    float*  proj   = (float*) alloc((size_t)LQ * NPROJ * 4);
    float*  qpts   = (float*) alloc((size_t)LQ * NH * PQ * 3 * 4);
    float*  kpts   = (float*) alloc((size_t)LQ * NH * PQ * 3 * 4);
    float*  vpts   = (float*) alloc((size_t)LQ * NH * PV * 3 * 4);
    float*  qq     = (float*) alloc((size_t)LQ * NH * 4);
    float*  kk     = (float*) alloc((size_t)LQ * NH * 4);
    __bf16* qa     = (__bf16*)alloc((size_t)NH * LQ * 32 * 2);
    __bf16* ka     = (__bf16*)alloc((size_t)NH * LQ * 32 * 2);
    float*  logits = (float*) alloc((size_t)NH * LQ * LQ * 4);
    __bf16* probs  = (__bf16*)alloc((size_t)NH * LQ * LQ * 2);
    float*  bsp    = (float*) alloc((size_t)LQ * NBR * NH * 4);
    __bf16* vcat   = (__bf16*)alloc((size_t)NH * NV * LQ * 2);
    float*  ovbuf  = (float*) alloc((size_t)NH * LQ * NV * 4);
    __bf16* catbf  = (__bf16*)alloc((size_t)LQ * NCAT * 2);
    __bf16* wout_t = (__bf16*)alloc((size_t)CS * NCAT * 2);

    auto grid1 = [](int n, int b) { return dim3((n + b - 1) / b); };

    // 1) bf16 copy of s, packed/transposed weights
    k_f2bf<<<grid1(LQ * CS, 256), 256, 0, stream>>>(s, s_bf, LQ * CS);
    k_pack_w<<<grid1(NPROJ * CS, 256), 256, 0, stream>>>(
        W_q, W_kv, W_qp, W_kvp, b_q, b_kv, b_qp, b_kvp, Wall_t, b_all);
    k_pack_wout<<<grid1(CS * NCAT, 256), 256, 0, stream>>>(W_out, wout_t);

    // 2) fused projection GEMM: proj = s @ [W_q|W_kv|W_qp|W_kvp] + biases
    {
        int tiles = (LQ / 32) * (NPROJ / 32);   // 24*36
        k_gemm_bf16<<<dim3((tiles + 3) / 4, 1), 128, 0, stream>>>(
            s_bf, Wall_t, proj, b_all, LQ, NPROJ, CS, 0, 0, 0);
    }

    // 3) rigid-transform points, point norms, augmented logit operands
    k_points<<<grid1(LQ * NH * (2 * PQ + PV), 256), 256, 0, stream>>>(
        proj, rot, trans, qpts, kpts, vpts);
    k_ptsq<<<grid1(LQ * NH, 256), 256, 0, stream>>>(qpts, kpts, qq, kk);
    k_build_aug<<<grid1(NH * LQ * 32, 256), 256, 0, stream>>>(
        proj, qpts, kpts, hwts, qa, ka);

    // 4) fused QK^T + point cross-term: batched per-head GEMM, K=32
    {
        int tiles = (LQ / 32) * (LQ / 32);      // 24*24
        k_gemm_bf16<<<dim3((tiles + 3) / 4, NH), 128, 0, stream>>>(
            qa, ka, logits, nullptr, LQ, LQ, 32,
            (long long)LQ * 32, (long long)LQ * 32, (long long)LQ * LQ);
    }

    // 5) sparse pair bias, logit completion, scatter, softmax
    k_bias_sp<<<grid1(LQ * NBR * NH, 256), 256, 0, stream>>>(z, W_b, bsp);
    k_finalize<<<grid1(NH * LQ * LQ, 256), 256, 0, stream>>>(
        logits, qq, kk, b_b, hwts, mask);
    k_scatter<<<grid1(LQ * NBR * NH, 256), 256, 0, stream>>>(logits, bsp, Eidx);
    k_softmax<<<dim3(NH * LQ), 256, 0, stream>>>(logits, probs);

    // 6) A @ [V | V_pts] : batched per-head GEMM, N=NV (padded), K=768
    k_build_vcat<<<grid1(NH * NV * LQ, 256), 256, 0, stream>>>(proj, vpts, vcat);
    {
        int tiles = (LQ / 32) * (NV / 32);      // 24*2
        k_gemm_bf16<<<dim3((tiles + 3) / 4, NH), 128, 0, stream>>>(
            probs, vcat, ovbuf, nullptr, LQ, NV, LQ,
            (long long)LQ * LQ, (long long)NV * LQ, (long long)LQ * NV);
    }

    // 7) sparse o_pair + cat assembly (o, inverse-transformed o_pt, norms)
    k_opair<<<dim3(LQ), 128, 0, stream>>>(probs, z, Eidx, catbf);
    k_fincat<<<dim3(LQ), 256, 0, stream>>>(ovbuf, rot, trans, catbf);

    // 8) output projection: out = cat @ W_out + b_out
    {
        int tiles = (LQ / 32) * (CS / 32);      // 24*12
        k_gemm_bf16<<<dim3((tiles + 3) / 4, 1), 128, 0, stream>>>(
            catbf, wout_t, out, b_out, LQ, CS, NCAT, 0, 0, 0);
    }
}
